// MoELayer_17325898072544
// MI455X (gfx1250) — matrix operand order repaired and checked by reading, not compiled, not run on hardware
//
#include <hip/hip_runtime.h>
#include <hip/hip_bf16.h>
#include <math.h>

typedef __bf16 bf16_t;
typedef __attribute__((ext_vector_type(16))) __bf16 v16bf;
typedef __attribute__((ext_vector_type(8)))  __bf16 v8bf;
typedef __attribute__((ext_vector_type(8)))  float  v8f;

// Problem sizes (fixed by the reference)
constexpr int Bn = 2, Tn = 2048, Dn = 1024, Hn = 4096, On = 1024, En = 8, Kn = 2;
constexpr int NT = Bn * Tn;       // 4096 tokens
constexpr int NE = NT * Kn;       // 8192 routed (token, expert) entries -- always exactly this many
constexpr int BM = 128, BN = 128, BK = 32;
constexpr int LDSP = 40;          // LDS row pitch in bf16 elements (80B: 16B-aligned, bank-spread)

// ---------------------------------------------------------------------------
// Router: one wave32 per token. logits = x@Wg+bg, noise_logits = x@Wn+bn,
// noisy = logits + noise*softplus(noise_logits); top-2; softmax over the two
// (masked softmax + L1 renorm over top-k == 2-way softmax).
// ---------------------------------------------------------------------------
__global__ __launch_bounds__(128) void router_kernel(
    const float* __restrict__ x, const float* __restrict__ noise,
    const float* __restrict__ Wg, const float* __restrict__ bg,
    const float* __restrict__ Wn, const float* __restrict__ bn,
    int* __restrict__ topk_idx, float* __restrict__ topk_gate)
{
  const int wave  = threadIdx.x >> 5;
  const int lane  = threadIdx.x & 31;
  const int token = blockIdx.x * 4 + wave;
  if (token >= NT) return;
  const float* xr = x + (size_t)token * Dn;

  float ag[En] = {}, an[En] = {};
  for (int d = lane; d < Dn; d += 32) {
    float xv = xr[d];
    const float* wgr = Wg + d * En;
    const float* wnr = Wn + d * En;
#pragma unroll
    for (int j = 0; j < En; ++j) {
      ag[j] = fmaf(xv, wgr[j], ag[j]);
      an[j] = fmaf(xv, wnr[j], an[j]);
    }
  }
#pragma unroll
  for (int j = 0; j < En; ++j) {
#pragma unroll
    for (int off = 16; off > 0; off >>= 1) {
      ag[j] += __shfl_xor(ag[j], off, 32);
      an[j] += __shfl_xor(an[j], off, 32);
    }
  }
  if (lane == 0) {
    float v0 = -INFINITY, v1 = -INFINITY; int i0 = 0, i1 = 0;
#pragma unroll
    for (int j = 0; j < En; ++j) {
      float lg = ag[j] + bg[j];
      float nl = an[j] + bn[j];
      float sp = (nl > 20.f) ? nl : log1pf(expf(nl));
      float nv = lg + noise[(size_t)token * En + j] * sp;
      if (nv > v0)      { v1 = v0; i1 = i0; v0 = nv; i0 = j; }
      else if (nv > v1) { v1 = nv; i1 = j; }
    }
    float t  = expf(v1 - v0);          // v0 >= v1, stable
    float g0 = 1.f / (1.f + t);
    float g1 = t   / (1.f + t);
    topk_idx[token * 2]     = i0;  topk_gate[token * 2]     = g0;
    topk_idx[token * 2 + 1] = i1;  topk_gate[token * 2 + 1] = g1;
  }
}

// ---------------------------------------------------------------------------
// Routing bookkeeping
// ---------------------------------------------------------------------------
__global__ void zero_counts_kernel(int* counts) {
  if (threadIdx.x < En) counts[threadIdx.x] = 0;
}
__global__ void hist_kernel(const int* __restrict__ topk_idx, int* __restrict__ counts) {
  int i = blockIdx.x * blockDim.x + threadIdx.x;
  if (i < NE) atomicAdd(&counts[topk_idx[i]], 1);
}
__global__ void scan_kernel(const int* __restrict__ counts, int* __restrict__ offsets,
                            int* __restrict__ fill) {
  if (threadIdx.x == 0) {
    int off = 0;
    for (int e = 0; e < En; ++e) { offsets[e] = off; fill[e] = off; off += counts[e]; }
    offsets[En] = off;
  }
}
// etok2[pos] = token*2 + slot  (slot = which of the token's K entries) -> gemm2
// writes partials without atomics and stays bitwise deterministic.
__global__ void scatter_kernel(const int* __restrict__ topk_idx,
                               const float* __restrict__ topk_gate,
                               int* __restrict__ fill,
                               int* __restrict__ etok2, float* __restrict__ egate) {
  int i = blockIdx.x * blockDim.x + threadIdx.x;
  if (i < NE) {
    int e   = topk_idx[i];
    int pos = atomicAdd(&fill[e], 1);
    etok2[pos] = i;              // == token*2 + slot
    egate[pos] = topk_gate[i];
  }
}

// ---------------------------------------------------------------------------
// Grouped GEMM1: h[entry, :] = relu( x[token(entry), :] @ W1[e] + b1[e] ), bf16 out.
// Block tile 128x128x(K=32); 8 waves, each wave owns a 32x64 subtile = 2x4 WMMA tiles.
// f32 global -> bf16 LDS conversion fused into staging. A row-major in LDS,
// B column-major in LDS so both WMMA fragments are contiguous 16B chunks that
// match the CDNA5 16-bit A/B VGPR layouts (CDNA5 ISA §7.12.2).
// ---------------------------------------------------------------------------
__global__ __launch_bounds__(256) void gemm1_kernel(
    const float* __restrict__ x, const float* __restrict__ W1, const float* __restrict__ b1,
    const int* __restrict__ counts, const int* __restrict__ offsets,
    const int* __restrict__ etok2, bf16_t* __restrict__ h)
{
  const int e    = blockIdx.z;
  const int cnt  = counts[e];
  const int mblk = blockIdx.x * BM;
  if (mblk >= cnt) return;                       // early-exit: data-dependent work, static grid
  const int m_base = offsets[e] + mblk;
  const int rows   = min(BM, cnt - mblk);
  const int n0     = blockIdx.y * BN;
  const float* __restrict__ W = W1 + (size_t)e * Dn * Hn;

  __shared__ bf16_t As[BM][LDSP];
  __shared__ bf16_t Bs[BN][LDSP];

  const int tid   = threadIdx.x;
  const int arow  = tid >> 1;                    // A loader: 2 threads per row
  const int ahalf = tid & 1;
  const float* xrow = nullptr;
  if (arow < rows) xrow = x + (size_t)(etok2[m_base + arow] >> 1) * Dn + ahalf * 16;
  const int bcol = tid & (BN - 1);               // B loader: thread per column, 2 k-groups
  const int bkg  = tid >> 7;

  float areg[16], breg[16];
  auto load_tiles = [&](int k0) {
    if (xrow) {
      const float4* p = (const float4*)(xrow + k0);
#pragma unroll
      for (int q = 0; q < 4; ++q) {
        float4 v = p[q];
        areg[4*q] = v.x; areg[4*q+1] = v.y; areg[4*q+2] = v.z; areg[4*q+3] = v.w;
      }
    } else {
#pragma unroll
      for (int q = 0; q < 16; ++q) areg[q] = 0.f;
    }
    const float* wp = W + (size_t)(k0 + bkg * 16) * Hn + n0 + bcol;   // coalesced over bcol
#pragma unroll
    for (int kk = 0; kk < 16; ++kk) breg[kk] = wp[(size_t)kk * Hn];
  };
  auto store_tiles = [&]() {
    v8bf pa0, pa1, pb0, pb1;
#pragma unroll
    for (int q = 0; q < 8; ++q) {
      pa0[q] = (bf16_t)areg[q];  pa1[q] = (bf16_t)areg[q + 8];
      pb0[q] = (bf16_t)breg[q];  pb1[q] = (bf16_t)breg[q + 8];
    }
    *(v8bf*)&As[arow][ahalf * 16]     = pa0;
    *(v8bf*)&As[arow][ahalf * 16 + 8] = pa1;
    *(v8bf*)&Bs[bcol][bkg * 16]       = pb0;   // column-major: k contiguous per column
    *(v8bf*)&Bs[bcol][bkg * 16 + 8]   = pb1;
  };

  const int lane = tid & 31;
  const int wv   = tid >> 5;
  const int wm   = wv & 3;            // 4 wave-rows * 32
  const int wn   = wv >> 2;           // 2 wave-cols * 64
  const int lrow = lane & 15;
  const int kh   = lane >> 4;

  v8f acc[2][4] = {};

  load_tiles(0);
  constexpr int KT = Dn / BK;
  for (int kt = 0; kt < KT; ++kt) {
    __syncthreads();
    store_tiles();
    __syncthreads();
    if (kt + 1 < KT) load_tiles((kt + 1) * BK);   // overlap next global loads with WMMA

    v16bf afrag[2], bfrag[4];
#pragma unroll
    for (int mi = 0; mi < 2; ++mi) {
      int r = wm * 32 + mi * 16 + lrow;
      v8bf lo = *(const v8bf*)&As[r][kh * 8];        // K = kh*8 .. +7
      v8bf hi = *(const v8bf*)&As[r][kh * 8 + 16];   // K = 16 + kh*8 .. +7
#pragma unroll
      for (int q = 0; q < 8; ++q) { afrag[mi][q] = lo[q]; afrag[mi][q + 8] = hi[q]; }
    }
#pragma unroll
    for (int ni = 0; ni < 4; ++ni) {
      int c = wn * 64 + ni * 16 + lrow;
      v8bf lo = *(const v8bf*)&Bs[c][kh * 8];        // K = kh*8 .. +7
      v8bf hi = *(const v8bf*)&Bs[c][kh * 8 + 16];   // K = 16 + kh*8 .. +7
#pragma unroll
      for (int q = 0; q < 8; ++q) { bfrag[ni][q] = lo[q]; bfrag[ni][q + 8] = hi[q]; }
    }
#pragma unroll
    for (int mi = 0; mi < 2; ++mi)
#pragma unroll
      for (int ni = 0; ni < 4; ++ni)
        acc[mi][ni] = __builtin_amdgcn_wmma_f32_16x16x32_bf16(
            false, afrag[mi], false, bfrag[ni], (short)0, acc[mi][ni], false, false);
  }

  const float* be = b1 + (size_t)e * Hn;
#pragma unroll
  for (int mi = 0; mi < 2; ++mi) {
#pragma unroll
    for (int ni = 0; ni < 4; ++ni) {
      int col = n0 + wn * 64 + ni * 16 + lrow;
      float bias = be[col];
#pragma unroll
      for (int r = 0; r < 8; ++r) {
        int rl = wm * 32 + mi * 16 + kh * 8 + r;     // C layout: VGPR r -> M = r + kh*8
        if (rl < rows) {
          float v = fmaxf(acc[mi][ni][r] + bias, 0.f);
          h[(size_t)(m_base + rl) * Hn + col] = (bf16_t)v;
        }
      }
    }
  }
}

// ---------------------------------------------------------------------------
// Grouped GEMM2: part[token*2+slot, :] = gate * ( h[entry,:] @ W2[e] + b2[e] )
// ---------------------------------------------------------------------------
__global__ __launch_bounds__(256) void gemm2_kernel(
    const bf16_t* __restrict__ h, const float* __restrict__ W2, const float* __restrict__ b2,
    const int* __restrict__ counts, const int* __restrict__ offsets,
    const int* __restrict__ etok2, const float* __restrict__ egate,
    float* __restrict__ part)
{
  const int e    = blockIdx.z;
  const int cnt  = counts[e];
  const int mblk = blockIdx.x * BM;
  if (mblk >= cnt) return;
  const int m_base = offsets[e] + mblk;
  const int rows   = min(BM, cnt - mblk);
  const int n0     = blockIdx.y * BN;
  const float* __restrict__ W = W2 + (size_t)e * Hn * On;

  __shared__ bf16_t As[BM][LDSP];
  __shared__ bf16_t Bs[BN][LDSP];

  const int tid   = threadIdx.x;
  const int arow  = tid >> 1;
  const int ahalf = tid & 1;
  const bf16_t* hrow = (arow < rows) ? h + (size_t)(m_base + arow) * Hn + ahalf * 16 : nullptr;
  const int bcol = tid & (BN - 1);
  const int bkg  = tid >> 7;

  v8bf areg0, areg1;
  float breg[16];
  auto load_tiles = [&](int k0) {
    if (hrow) {
      areg0 = *(const v8bf*)(hrow + k0);
      areg1 = *(const v8bf*)(hrow + k0 + 8);
    } else {
#pragma unroll
      for (int q = 0; q < 8; ++q) { areg0[q] = (bf16_t)0.f; areg1[q] = (bf16_t)0.f; }
    }
    const float* wp = W + (size_t)(k0 + bkg * 16) * On + n0 + bcol;
#pragma unroll
    for (int kk = 0; kk < 16; ++kk) breg[kk] = wp[(size_t)kk * On];
  };
  auto store_tiles = [&]() {
    v8bf pb0, pb1;
#pragma unroll
    for (int q = 0; q < 8; ++q) { pb0[q] = (bf16_t)breg[q]; pb1[q] = (bf16_t)breg[q + 8]; }
    *(v8bf*)&As[arow][ahalf * 16]     = areg0;
    *(v8bf*)&As[arow][ahalf * 16 + 8] = areg1;
    *(v8bf*)&Bs[bcol][bkg * 16]       = pb0;
    *(v8bf*)&Bs[bcol][bkg * 16 + 8]   = pb1;
  };

  const int lane = tid & 31;
  const int wv   = tid >> 5;
  const int wm   = wv & 3;
  const int wn   = wv >> 2;
  const int lrow = lane & 15;
  const int kh   = lane >> 4;

  v8f acc[2][4] = {};

  load_tiles(0);
  constexpr int KT = Hn / BK;
  for (int kt = 0; kt < KT; ++kt) {
    __syncthreads();
    store_tiles();
    __syncthreads();
    if (kt + 1 < KT) load_tiles((kt + 1) * BK);

    v16bf afrag[2], bfrag[4];
#pragma unroll
    for (int mi = 0; mi < 2; ++mi) {
      int r = wm * 32 + mi * 16 + lrow;
      v8bf lo = *(const v8bf*)&As[r][kh * 8];
      v8bf hi = *(const v8bf*)&As[r][kh * 8 + 16];
#pragma unroll
      for (int q = 0; q < 8; ++q) { afrag[mi][q] = lo[q]; afrag[mi][q + 8] = hi[q]; }
    }
#pragma unroll
    for (int ni = 0; ni < 4; ++ni) {
      int c = wn * 64 + ni * 16 + lrow;
      v8bf lo = *(const v8bf*)&Bs[c][kh * 8];
      v8bf hi = *(const v8bf*)&Bs[c][kh * 8 + 16];
#pragma unroll
      for (int q = 0; q < 8; ++q) { bfrag[ni][q] = lo[q]; bfrag[ni][q + 8] = hi[q]; }
    }
#pragma unroll
    for (int mi = 0; mi < 2; ++mi)
#pragma unroll
      for (int ni = 0; ni < 4; ++ni)
        acc[mi][ni] = __builtin_amdgcn_wmma_f32_16x16x32_bf16(
            false, afrag[mi], false, bfrag[ni], (short)0, acc[mi][ni], false, false);
  }

  const float* be = b2 + (size_t)e * On;
#pragma unroll
  for (int mi = 0; mi < 2; ++mi) {
#pragma unroll
    for (int ni = 0; ni < 4; ++ni) {
      int col = n0 + wn * 64 + ni * 16 + lrow;
      float bias = be[col];
#pragma unroll
      for (int r = 0; r < 8; ++r) {
        int rl = wm * 32 + mi * 16 + kh * 8 + r;
        if (rl < rows) {
          int   entry = m_base + rl;
          int   slot  = etok2[entry];                 // token*2 + k-slot: unique -> no atomics
          float g     = egate[entry];
          part[(size_t)slot * On + col] = g * (acc[mi][ni][r] + bias);
        }
      }
    }
  }
}

// out[t, :] = part[2t, :] + part[2t+1, :]
__global__ void combine_kernel(const float4* __restrict__ part4, float4* __restrict__ out4) {
  int i = blockIdx.x * blockDim.x + threadIdx.x;        // over NT * On/4
  constexpr int C4 = On / 4;
  if (i < NT * C4) {
    int t = i / C4, c = i % C4;
    float4 a = part4[(size_t)(2 * t) * C4 + c];
    float4 b = part4[(size_t)(2 * t + 1) * C4 + c];
    out4[i] = make_float4(a.x + b.x, a.y + b.y, a.z + b.z, a.w + b.w);
  }
}

// ---------------------------------------------------------------------------
extern "C" void kernel_launch(void* const* d_in, const int* in_sizes, int n_in,
                              void* d_out, int out_size, void* d_ws, size_t ws_size,
                              hipStream_t stream) {
  (void)in_sizes; (void)n_in; (void)out_size; (void)ws_size;
  const float* x     = (const float*)d_in[0];
  const float* noise = (const float*)d_in[1];
  const float* Wg    = (const float*)d_in[2];
  const float* bg    = (const float*)d_in[3];
  const float* Wn    = (const float*)d_in[4];
  const float* bn    = (const float*)d_in[5];
  const float* W1    = (const float*)d_in[6];
  const float* b1    = (const float*)d_in[7];
  const float* W2    = (const float*)d_in[8];
  const float* b2    = (const float*)d_in[9];
  float* out = (float*)d_out;

  // Workspace layout (256B-aligned blocks)
  uint8_t* ws = (uint8_t*)d_ws;
  int*    counts    = (int*)  (ws + 0);
  int*    offsets   = (int*)  (ws + 256);
  int*    fill      = (int*)  (ws + 512);
  int*    topk_idx  = (int*)  (ws + 1024);                         // NE*4   = 32 KB
  float*  topk_gate = (float*)(ws + 1024 + 32768);                 // 32 KB
  int*    etok2     = (int*)  (ws + 1024 + 65536);                 // 32 KB
  float*  egate     = (float*)(ws + 1024 + 98304);                 // 32 KB
  bf16_t* hbuf      = (bf16_t*)(ws + 132096);                      // NE*Hn*2 = 64 MB
  float*  part      = (float*) (ws + 132096 + (size_t)NE * Hn * 2); // NE*On*4 = 32 MB

  zero_counts_kernel<<<1, 32, 0, stream>>>(counts);
  router_kernel<<<NT / 4, 128, 0, stream>>>(x, noise, Wg, bg, Wn, bn, topk_idx, topk_gate);
  hist_kernel<<<(NE + 255) / 256, 256, 0, stream>>>(topk_idx, counts);
  scan_kernel<<<1, 32, 0, stream>>>(counts, offsets, fill);
  scatter_kernel<<<(NE + 255) / 256, 256, 0, stream>>>(topk_idx, topk_gate, fill, etok2, egate);

  dim3 g1(NT / BM, Hn / BN, En);   // 32 x 32 x 8, blocks past counts[e] early-exit
  gemm1_kernel<<<g1, 256, 0, stream>>>(x, W1, b1, counts, offsets, etok2, hbuf);
  dim3 g2(NT / BM, On / BN, En);   // 32 x 8 x 8
  gemm2_kernel<<<g2, 256, 0, stream>>>(hbuf, W2, b2, counts, offsets, etok2, egate, part);
  combine_kernel<<<(NT * On / 4 + 255) / 256, 256, 0, stream>>>((const float4*)part, (float4*)out);
}
